// SRUppModel_26319559590339
// MI455X (gfx1250) — matrix operand order repaired and checked by reading, not compiled, not run on hardware
//
#include <hip/hip_runtime.h>
#include <hip/hip_bf16.h>

// ---------------------------------------------------------------------------
// SRU++ forward for MI455X (gfx1250).  All heavy einsums run through
// v_wmma_f32_16x16x32_bf16 (wave32 WMMA), fp32 accumulation.  Softmax stages
// score rows into LDS with gfx1250 async global->LDS loads.
// ---------------------------------------------------------------------------

typedef __attribute__((ext_vector_type(16))) __bf16 bf16x16;
typedef __attribute__((ext_vector_type(8)))  __bf16 bf16x8;
typedef __attribute__((ext_vector_type(4)))  __bf16 bf16x4;
typedef __attribute__((ext_vector_type(8)))  float  floatx8;

__device__ __forceinline__ __bf16 f2bf(float x) {
    union { float f; unsigned u; } a; a.f = x;
    unsigned r = (a.u + 0x7FFFu + ((a.u >> 16) & 1u)) >> 16;  // RNE
    union { unsigned short s; __bf16 b; } o; o.s = (unsigned short)r;
    return o.b;
}

__device__ __forceinline__ bf16x4 f2bf4(float x, float y, float z, float w) {
    bf16x4 t; t.x = f2bf(x); t.y = f2bf(y); t.z = f2bf(z); t.w = f2bf(w);
    return t;
}

// ---------------------------------------------------------------------------
// Generalized batched GEMM:  C[m,n] = outScale * sum_k A[m,k] * B[k,n]
//   A addr: aBase + batch*aBatch + m*aM + k          (k contiguous)
//   B addr: bBase + batch*bBatch + k*bK + n*bN       (bN==1 or bK==1 fast paths)
//   C addr: cBase + batch*cBatch + m*cM + n
// Block tile 128x128, 8 waves (2x4), each wave 64x32 = 4x2 WMMA tiles, K step 32.
// All fast-path global accesses are 16B aligned by construction of the call sites.
// ---------------------------------------------------------------------------
__global__ __launch_bounds__(256)
void gemm_bf16_wmma(const float* __restrict__ A, long aM, long aBatch,
                    const float* __restrict__ Bm, long bK, long bN, long bBatch,
                    float* __restrict__ C, long cM, long cBatch,
                    int M, int N, int K, float outScale) {
    __shared__ __bf16 ldsA[128][32];   // [m][k]  row-major
    __shared__ __bf16 ldsBt[128][32];  // [n][k]  B stored transposed

    const int tid   = threadIdx.x;
    const int lane  = tid & 31;
    const int waveId = tid >> 5;
    const int waveM = waveId >> 2;       // 0..1  -> 64 rows each
    const int waveN = waveId & 3;        // 0..3  -> 32 cols each
    const int halfSel = lane >> 4;       // which K-half this lane holds
    const int rc      = lane & 15;       // row (A) / col (B,C) within tile

    const long batch = blockIdx.z;
    A  += batch * aBatch;
    Bm += batch * bBatch;
    C  += batch * cBatch;

    const int blockM = blockIdx.y * 128;
    const int blockN = blockIdx.x * 128;

    const bool fullM = (blockM + 128 <= M);
    const bool fullN = (blockN + 128 <= N);

    floatx8 acc[4][2] = {};

    for (int k0 = 0; k0 < K; k0 += 32) {
        const bool fullK = (k0 + 32 <= K);
        __syncthreads();

        // ---- stage A tile: 128 x 32 ----
        if (fullM && fullK) {                       // block-uniform fast path
            #pragma unroll
            for (int i = 0; i < 4; ++i) {
                int r = (tid >> 3) + i * 32;
                int c = (tid & 7) * 4;
                const float* pa = A + (long)(blockM + r) * aM + (k0 + c);
                const float4 f = *(const float4*)pa;
                if (k0 + 32 < K) __builtin_prefetch(pa + 32, 0, 1);
                *(bf16x4*)&ldsA[r][c] = f2bf4(f.x, f.y, f.z, f.w);
            }
        } else {
            #pragma unroll
            for (int i = 0; i < 4; ++i) {
                int r = (tid >> 3) + i * 32;
                int c = (tid & 7) * 4;
                long gm = blockM + r;
                long gk = k0 + c;
                float v0 = (gm < M && gk + 0 < K) ? A[gm * aM + gk + 0] : 0.0f;
                float v1 = (gm < M && gk + 1 < K) ? A[gm * aM + gk + 1] : 0.0f;
                float v2 = (gm < M && gk + 2 < K) ? A[gm * aM + gk + 2] : 0.0f;
                float v3 = (gm < M && gk + 3 < K) ? A[gm * aM + gk + 3] : 0.0f;
                *(bf16x4*)&ldsA[r][c] = f2bf4(v0, v1, v2, v3);
            }
        }

        // ---- stage B tile: 32 x 128, transposed into ldsBt[n][k] ----
        if (fullN && fullK && bN == 1) {
            // contiguous along n: vector load, scatter-transpose into LDS
            #pragma unroll
            for (int i = 0; i < 4; ++i) {
                int kk = (tid >> 5) + i * 8;
                int nc = (tid & 31) * 4;
                const float4 f = *(const float4*)(Bm + (long)(k0 + kk) * bK + (blockN + nc));
                ldsBt[nc + 0][kk] = f2bf(f.x);
                ldsBt[nc + 1][kk] = f2bf(f.y);
                ldsBt[nc + 2][kk] = f2bf(f.z);
                ldsBt[nc + 3][kk] = f2bf(f.w);
            }
        } else if (fullN && fullK && bK == 1) {
            // contiguous along k: vector load AND vector LDS store
            #pragma unroll
            for (int i = 0; i < 4; ++i) {
                int nn = (tid >> 3) + i * 32;
                int kc = (tid & 7) * 4;
                const float4 f = *(const float4*)(Bm + (long)(blockN + nn) * bN + (k0 + kc));
                *(bf16x4*)&ldsBt[nn][kc] = f2bf4(f.x, f.y, f.z, f.w);
            }
        } else {
            #pragma unroll
            for (int j = 0; j < 16; ++j) {
                int idx = tid + j * 256;      // 0..4095
                int kk = idx >> 7;            // 0..31
                int nn = idx & 127;           // 0..127
                long gk = k0 + kk;
                long gn = blockN + nn;
                float v = (gk < K && gn < N) ? Bm[gk * bK + gn * bN] : 0.0f;
                ldsBt[nn][kk] = f2bf(v);
            }
        }
        __syncthreads();

        // ---- fragments per ISA layout ----
        // A 16x32 bf16: lane(0-15)=row m, holds K[h*8..h*8+7] then K[16+h*8..+7]
        bf16x16 afrag[4];
        #pragma unroll
        for (int mi = 0; mi < 4; ++mi) {
            const __bf16* pa = &ldsA[waveM * 64 + mi * 16 + rc][halfSel * 8];
            union { bf16x16 v; bf16x8 h[2]; } u;
            u.h[0] = *(const bf16x8*)(pa);
            u.h[1] = *(const bf16x8*)(pa + 16);
            afrag[mi] = u.v;
        }
        // B 32x16 bf16: lane holds col n, K[h*8..h*8+7] then K[16+h*8..+7]
        bf16x16 bfrag[2];
        #pragma unroll
        for (int ni = 0; ni < 2; ++ni) {
            const __bf16* pb = &ldsBt[waveN * 32 + ni * 16 + rc][halfSel * 8];
            union { bf16x16 v; bf16x8 h[2]; } u;
            u.h[0] = *(const bf16x8*)(pb);
            u.h[1] = *(const bf16x8*)(pb + 16);
            bfrag[ni] = u.v;
        }

        #pragma unroll
        for (int mi = 0; mi < 4; ++mi)
            #pragma unroll
            for (int ni = 0; ni < 2; ++ni)
                acc[mi][ni] = __builtin_amdgcn_wmma_f32_16x16x32_bf16(
                    false, afrag[mi], false, bfrag[ni],
                    (short)0, acc[mi][ni], false, false);
    }

    // ---- store C: f32 16x16 layout: VGPR v -> row halfSel*8+v, col = rc ----
    #pragma unroll
    for (int mi = 0; mi < 4; ++mi)
        #pragma unroll
        for (int ni = 0; ni < 2; ++ni)
            #pragma unroll
            for (int v = 0; v < 8; ++v) {
                long row = blockM + waveM * 64 + mi * 16 + halfSel * 8 + v;
                long col = blockN + waveN * 32 + ni * 16 + rc;
                if (row < (long)M && col < (long)N)
                    C[row * cM + col] = outScale * acc[mi][ni][v];
            }
}

// ---------------------------------------------------------------------------
// Row softmax, rowLen == 1024, one block of 256 threads per row.
// Row is staged into LDS with gfx1250 async global->LDS loads (ASYNCcnt),
// then all three passes (max / exp-sum / normalize) run out of LDS.
// ---------------------------------------------------------------------------
__global__ __launch_bounds__(256)
void softmax_rows(float* __restrict__ s, int rowLen) {
    float* p = s + (long)blockIdx.x * rowLen;
    __shared__ float rowbuf[1024];
    __shared__ float red[256];
    const int tid = threadIdx.x;

    // stage row -> LDS: one 16B chunk per thread (256 * 16B = 4KB row)
    {
        const float* g = p + tid * 4;
#if defined(__gfx1250__)
        // generic pointer to LDS: low 32 bits are the LDS byte address
        unsigned loff = (unsigned)(unsigned long long)&rowbuf[tid * 4];
        asm volatile("global_load_async_to_lds_b128 %0, %1, off"
                     :: "v"(loff), "v"(g) : "memory");
        asm volatile("s_wait_asynccnt 0x0" ::: "memory");
#else
        rowbuf[tid * 4 + 0] = g[0];
        rowbuf[tid * 4 + 1] = g[1];
        rowbuf[tid * 4 + 2] = g[2];
        rowbuf[tid * 4 + 3] = g[3];
#endif
    }
    __syncthreads();

    float m = -1e30f;
    for (int i = tid; i < rowLen; i += 256) m = fmaxf(m, rowbuf[i]);
    red[tid] = m; __syncthreads();
    for (int off = 128; off > 0; off >>= 1) {
        if (tid < off) red[tid] = fmaxf(red[tid], red[tid + off]);
        __syncthreads();
    }
    m = red[0]; __syncthreads();

    float sum = 0.0f;
    for (int i = tid; i < rowLen; i += 256) {
        float e = __expf(rowbuf[i] - m);
        rowbuf[i] = e;
        sum += e;
    }
    red[tid] = sum; __syncthreads();
    for (int off = 128; off > 0; off >>= 1) {
        if (tid < off) red[tid] += red[tid + off];
        __syncthreads();
    }
    float inv = 1.0f / red[0];
    for (int i = tid; i < rowLen; i += 256) p[i] = rowbuf[i] * inv;
}

// attnz = alpha * attn + z   (alpha read from device scalar)
__global__ __launch_bounds__(256)
void axpy_residual(const float* __restrict__ attn, const float* __restrict__ z,
                   const float* __restrict__ alpha, float* __restrict__ out, long n) {
    long i = (long)blockIdx.x * 256 + threadIdx.x;
    if (i < n) out[i] = alpha[0] * attn[i] + z[i];
}

// SRU recurrence: independent per (b,h); sequential over L.
__global__ __launch_bounds__(256)
void sru_scan(const float* __restrict__ U, const float* __restrict__ x,
              const float* __restrict__ vf, const float* __restrict__ vr,
              const float* __restrict__ bf, const float* __restrict__ br,
              float* __restrict__ h, int Lp, int Bp, int Hp, int Dp) {
    int idx = blockIdx.x * blockDim.x + threadIdx.x;
    if (idx >= Bp * Hp) return;
    int b  = idx / Hp;
    int hh = idx % Hp;
    const float vfv = vf[hh], vrv = vr[hh], bfv = bf[hh], brv = br[hh];
    const float SCALE_X = 1.41421356237309515f;  // sqrt(1 + exp(0))
    float c = 0.0f;
    for (int l = 0; l < Lp; ++l) {
        long lb = (long)l * Bp + b;
        long ub = lb * 3 * Hp + hh;
        float u0 = U[ub];
        float u1 = U[ub + Hp];
        float u2 = U[ub + 2 * Hp];
        float xt = x[lb * Dp + hh];
        float f  = 1.0f / (1.0f + __expf(-(u1 + vfv * c + bfv)));
        float cn = f * c + (1.0f - f) * u0;
        float r  = 1.0f / (1.0f + __expf(-(u2 + vrv * c + brv)));
        h[lb * Hp + hh] = r * cn + (1.0f - r) * xt * SCALE_X;
        c = cn;
    }
}

// out[b,c] = h_last[b,:] . lin_w[:,c] + lin_b[c]   (32 x 2)
__global__ __launch_bounds__(64)
void final_linear(const float* __restrict__ hlast, const float* __restrict__ w,
                  const float* __restrict__ bias, float* __restrict__ out,
                  int Bp, int Hp, int Cp) {
    int idx = blockIdx.x * blockDim.x + threadIdx.x;
    if (idx >= Bp * Cp) return;
    int b = idx / Cp, c = idx % Cp;
    float s = bias[c];
    for (int hh = 0; hh < Hp; ++hh) s += hlast[(long)b * Hp + hh] * w[(long)hh * Cp + c];
    out[idx] = s;
}

// ---------------------------------------------------------------------------
// Host orchestration
// ---------------------------------------------------------------------------
extern "C" void kernel_launch(void* const* d_in, const int* in_sizes, int n_in,
                              void* d_out, int out_size, void* d_ws, size_t ws_size,
                              hipStream_t stream) {
    constexpr int L_ = 1024, B_ = 32, D_ = 512, H_ = 512, P_ = 784, C_ = 2;
    constexpr int NLAYERS = 2;
    constexpr float Q_SCALE = 0.035714285714285712f;  // 784^-0.5

    constexpr long MROWS = (long)L_ * B_;       // 32768
    constexpr long ZSZ   = MROWS * P_;          // z / attn size
    constexpr long KVSZ  = 2 * ZSZ;             // kv size (>= U size = MROWS*3H)
    constexpr long SSZ   = (long)B_ * L_ * L_;  // scores (>= ZSZ, reused for attnz)
    constexpr long HSZ   = MROWS * H_;

    float* ws      = (float*)d_ws;
    float* zbuf    = ws;
    float* kvbuf   = zbuf + ZSZ;      // also reused as U buffer
    float* sbuf    = kvbuf + KVSZ;    // also reused as attnz buffer
    float* attnbuf = sbuf + SSZ;
    float* hbuf1   = attnbuf + ZSZ;
    float* hbuf2   = hbuf1 + HSZ;
    (void)ws_size; (void)in_sizes; (void)n_in; (void)out_size;

    const float* x = (const float*)d_in[0];

    const float* layer_in = x;
    float* layer_out[2] = { hbuf1, hbuf2 };

    dim3 blk(256);

    for (int layer = 0; layer < NLAYERS; ++layer) {
        const int base = 1 + layer * 8;
        const float* W1    = (const float*)d_in[base + 0];   // (512, 784)
        const float* W2    = (const float*)d_in[base + 1];   // (784, 1568)
        const float* W3    = (const float*)d_in[base + 2];   // (784, 1536)
        const float* alpha = (const float*)d_in[base + 3];   // scalar
        const float* vf    = (const float*)d_in[base + 4];
        const float* vr    = (const float*)d_in[base + 5];
        const float* bfp   = (const float*)d_in[base + 6];
        const float* brp   = (const float*)d_in[base + 7];

        // 1) z = layer_in(32768x512) @ W1(512x784)
        {
            dim3 grid((P_ + 127) / 128, (MROWS + 127) / 128, 1);
            gemm_bf16_wmma<<<grid, blk, 0, stream>>>(
                layer_in, (long)D_, 0L,
                W1, (long)P_, 1L, 0L,
                zbuf, (long)P_, 0L,
                (int)MROWS, P_, D_, 1.0f);
        }
        // 2) kv = z(32768x784) @ W2(784x1568)
        {
            dim3 grid((2 * P_ + 127) / 128, (MROWS + 127) / 128, 1);
            gemm_bf16_wmma<<<grid, blk, 0, stream>>>(
                zbuf, (long)P_, 0L,
                W2, (long)(2 * P_), 1L, 0L,
                kvbuf, (long)(2 * P_), 0L,
                (int)MROWS, 2 * P_, P_, 1.0f);
        }
        // 3) S[b] = Q_SCALE * Z_b(LxP) @ K_b^T(PxL)   -- batched over B
        //    A[m=l][k=p] = z[(l*B+b)*P + p]       (aM = B*P, batch off = b*P)
        //    B[k=p][n=m] = kv[(m*B+b)*2P + p]     (bK = 1, bN = B*2P, batch off = b*2P)
        {
            dim3 grid((L_ + 127) / 128, (L_ + 127) / 128, B_);
            gemm_bf16_wmma<<<grid, blk, 0, stream>>>(
                zbuf, (long)B_ * P_, (long)P_,
                kvbuf, 1L, (long)B_ * 2 * P_, (long)(2 * P_),
                sbuf, (long)L_, (long)L_ * L_,
                L_, L_, P_, Q_SCALE);
        }
        // 4) softmax over m (rows of length L)
        softmax_rows<<<dim3(B_ * L_), blk, 0, stream>>>(sbuf, L_);
        // 5) attn[l,b,p] = sum_m A[b,l,m] * v[m,b,p]   -- batched over B
        //    B[k=m][n=p] = kv[(m*B+b)*2P + P + p]  (bK = B*2P, bN = 1)
        {
            dim3 grid((P_ + 127) / 128, (L_ + 127) / 128, B_);
            gemm_bf16_wmma<<<grid, blk, 0, stream>>>(
                sbuf, (long)L_, (long)L_ * L_,
                kvbuf + P_, (long)B_ * 2 * P_, 1L, (long)(2 * P_),
                attnbuf, (long)B_ * P_, (long)P_,
                L_, P_, L_, 1.0f);
        }
        // 6) attnz = alpha*attn + z   (into sbuf; scores are dead now)
        axpy_residual<<<dim3((unsigned)((ZSZ + 255) / 256)), blk, 0, stream>>>(
            attnbuf, zbuf, alpha, sbuf, ZSZ);
        // 7) U = attnz(32768x784) @ W3(784x1536)   (into kvbuf; kv is dead now)
        {
            dim3 grid((3 * H_ + 127) / 128, (MROWS + 127) / 128, 1);
            gemm_bf16_wmma<<<grid, blk, 0, stream>>>(
                sbuf, (long)P_, 0L,
                W3, (long)(3 * H_), 1L, 0L,
                kvbuf, (long)(3 * H_), 0L,
                (int)MROWS, 3 * H_, P_, 1.0f);
        }
        // 8) recurrence over L
        sru_scan<<<dim3((B_ * H_ + 255) / 256), blk, 0, stream>>>(
            kvbuf, layer_in, vf, vr, bfp, brp, layer_out[layer],
            L_, B_, H_, D_);

        layer_in = layer_out[layer];
    }

    // final: out = h[-1] @ lin_w + lin_b
    const float* lin_w = (const float*)d_in[1 + NLAYERS * 8];
    const float* lin_b = (const float*)d_in[1 + NLAYERS * 8 + 1];
    const float* hlast = hbuf2 + (long)(L_ - 1) * B_ * H_;
    final_linear<<<dim3(1), dim3(64), 0, stream>>>(
        hlast, lin_w, lin_b, (float*)d_out, B_, H_, C_);
}
